// BinaryLinear_60455959658593
// MI455X (gfx1250) — 8-bit matrix operand order repaired and checked by reading, compiled, not run on hardware
//
#include <hip/hip_runtime.h>

typedef __attribute__((ext_vector_type(16))) int   v16i;
typedef __attribute__((ext_vector_type(4)))  int   v4i;
typedef __attribute__((ext_vector_type(8)))  float v8f;

#define MDIM 8192
#define NDIM 4096
#define KDIM 4096
#define TBM 128
#define TBN 128
#define TBK 128
#define LDSS 144

#if __has_builtin(__builtin_amdgcn_global_load_async_to_lds_b128)
#define HAVE_ASYNC_LDS 1
typedef int v4i_n __attribute__((vector_size(16)));
typedef __attribute__((address_space(1))) v4i_n* gptr128_t;
typedef __attribute__((address_space(3))) v4i_n* lptr128_t;
#else
#define HAVE_ASYNC_LDS 0
#endif

#if HAVE_ASYNC_LDS
static __device__ __forceinline__ void wait_asynccnt0() {
#if __has_builtin(__builtin_amdgcn_s_wait_asynccnt)
  __builtin_amdgcn_s_wait_asynccnt(0);
#else
  asm volatile("s_wait_asynccnt 0x0" ::: "memory");
#endif
}
#endif

__global__ __launch_bounds__(256) void pack_sign_fp8_kernel(
    const float* __restrict__ in, unsigned int* __restrict__ out, int n4) {
  int i = blockIdx.x * blockDim.x + threadIdx.x;
  if (i >= n4) return;
  float4 v = ((const float4*)in)[i];
  unsigned b0 = (v.x > 0.0f) ? 0x38u : ((v.x < 0.0f) ? 0xB8u : 0x00u);
  unsigned b1 = (v.y > 0.0f) ? 0x38u : ((v.y < 0.0f) ? 0xB8u : 0x00u);
  unsigned b2 = (v.z > 0.0f) ? 0x38u : ((v.z < 0.0f) ? 0xB8u : 0x00u);
  unsigned b3 = (v.w > 0.0f) ? 0x38u : ((v.w < 0.0f) ? 0xB8u : 0x00u);
  out[i] = b0 | (b1 << 8) | (b2 << 16) | (b3 << 24);
}

__global__ __launch_bounds__(256) void bgemm_fp8_wmma(
    const unsigned char* __restrict__ A,
    const unsigned char* __restrict__ B,
    const float* __restrict__ bias,
    float* __restrict__ out)
{
  __shared__ __align__(16) unsigned char lds[2][2][TBM * LDSS];

  const int tid  = threadIdx.x;
  const int lane = tid & 31;
  const int wv   = tid >> 5;
  const int wm   = wv & 3;
  const int wn   = wv >> 2;
  const int bm   = blockIdx.y * TBM;
  const int bn   = blockIdx.x * TBN;

  int srow[4], scol[4];
#pragma unroll
  for (int r = 0; r < 4; ++r) {
    const int chunk = tid + r * 256;
    srow[r] = chunk >> 3;
    scol[r] = (chunk & 7) << 4;
  }

  v8f acc[2][4];
#pragma unroll
  for (int mt = 0; mt < 2; ++mt)
#pragma unroll
    for (int nt = 0; nt < 4; ++nt)
#pragma unroll
      for (int i = 0; i < 8; ++i) acc[mt][nt][i] = 0.0f;

  const int KT = KDIM / TBK;

  auto stage_tile = [&](int buf, int kt) {
    const size_t k0 = (size_t)kt * TBK;
#pragma unroll
    for (int r = 0; r < 4; ++r) {
      const unsigned char* ga = A + (size_t)(bm + srow[r]) * KDIM + k0 + scol[r];
      const unsigned char* gb = B + (size_t)(bn + srow[r]) * KDIM + k0 + scol[r];
      unsigned char* la = &lds[buf][0][srow[r] * LDSS + scol[r]];
      unsigned char* lb = &lds[buf][1][srow[r] * LDSS + scol[r]];
#if HAVE_ASYNC_LDS
      __builtin_amdgcn_global_load_async_to_lds_b128(
          (gptr128_t)(unsigned char*)ga, (lptr128_t)la, 0, 0);
      __builtin_amdgcn_global_load_async_to_lds_b128(
          (gptr128_t)(unsigned char*)gb, (lptr128_t)lb, 0, 0);
#else
      int4 av = *(const int4*)ga;
      int4 bv = *(const int4*)gb;
      *(int4*)la = av;
      *(int4*)lb = bv;
#endif
    }
  };

  stage_tile(0, 0);
#if HAVE_ASYNC_LDS
  wait_asynccnt0();
#endif
  __syncthreads();

  for (int kt = 0; kt < KT; ++kt) {
    const int  cur  = kt & 1;
    const int  nxt  = cur ^ 1;
    const bool more = (kt + 1) < KT;

    if (more) stage_tile(nxt, kt + 1);

    {
      const unsigned char* la = &lds[cur][0][0];
      const unsigned char* lb = &lds[cur][1][0];
      const int mr = lane & 15;
      const int g  = lane >> 4;

      union AU { unsigned long long q[8]; v16i v; } af[2];
#pragma unroll
      for (int mt = 0; mt < 2; ++mt) {
        const int base = (wm * 32 + mt * 16 + mr) * LDSS;
#pragma unroll
        for (int h = 0; h < 2; ++h)
#pragma unroll
          for (int p = 0; p < 4; ++p)
            af[mt].q[h * 4 + p] =
                *(const unsigned long long*)(la + base + h * 64 + p * 16 + g * 8);
      }

#pragma unroll
      for (int nt = 0; nt < 4; ++nt) {
        union BU { v4i h[4]; unsigned long long q[8]; v16i v; } bf;
        const int base = (wn * 64 + nt * 16 + mr) * LDSS + g * 8;
#pragma unroll
        for (int q = 0; q < 8; ++q)
          bf.q[q] = *(const unsigned long long*)(lb + base + (q >> 2) * 64 + (q & 3) * 16);

        acc[0][nt] = __builtin_amdgcn_wmma_f32_16x16x128_fp8_fp8(
            af[0].v, bf.v, (short)0, acc[0][nt], false, false);
        acc[1][nt] = __builtin_amdgcn_wmma_f32_16x16x128_fp8_fp8(
            af[1].v, bf.v, (short)0, acc[1][nt], false, false);
      }
    }

#if HAVE_ASYNC_LDS
    if (more) wait_asynccnt0();
#endif
    __syncthreads();
  }

#pragma unroll
  for (int nt = 0; nt < 4; ++nt) {
    const int   col = bn + wn * 64 + nt * 16 + (lane & 15);
    const float bv  = bias[col];
#pragma unroll
    for (int mt = 0; mt < 2; ++mt) {
      const int rbase = bm + wm * 32 + mt * 16 + (lane >> 4) * 8;
#pragma unroll
      for (int i = 0; i < 8; ++i)
        __builtin_nontemporal_store(acc[mt][nt][i] + bv,
                                    &out[(size_t)(rbase + i) * NDIM + col]);
    }
  }
}

extern "C" void kernel_launch(void* const* d_in, const int* in_sizes, int n_in,
                              void* d_out, int out_size, void* d_ws, size_t ws_size,
                              hipStream_t stream) {
  const float* x    = (const float*)d_in[0];
  const float* w    = (const float*)d_in[1];
  const float* bias = (const float*)d_in[2];
  float*       out  = (float*)d_out;

  unsigned char* sx = (unsigned char*)d_ws;
  unsigned char* sw = sx + (size_t)MDIM * KDIM;

  const int nx4 = (MDIM * KDIM) / 4;
  const int nw4 = (NDIM * KDIM) / 4;
  pack_sign_fp8_kernel<<<nx4 / 256, 256, 0, stream>>>(x, (unsigned int*)sx, nx4);
  pack_sign_fp8_kernel<<<nw4 / 256, 256, 0, stream>>>(w, (unsigned int*)sw, nw4);

  dim3 grid(NDIM / TBN, MDIM / TBM);
  bgemm_fp8_wmma<<<grid, 256, 0, stream>>>(sx, sw, bias, out);
}
